// EGCN_47287589929285
// MI455X (gfx1250) — matrix operand order repaired and checked by reading, not compiled, not run on hardware
//
#include <hip/hip_runtime.h>
#include <math.h>

// ---------------------------------------------------------------------------
// EvolveGCN: T=32 steps of {topk-embed -> GRU weight evolution -> GCN}.
// Phase 1 (sequential, 1 block/step, 1024 thr): scores + top-64 (bitonic) +
//   GRU via v_wmma_f32_16x16x4_f32 -> Qn_all[t].
// Phase 2 (parallel): Y[t] = Nodes[t] @ Qn[t], stored bf16.
// Phase 3 (parallel, HBM-bound): out[t] = relu(A[t] @ Y[t]) via
//   v_wmma_f32_16x16x32_bf16; A streamed fp32 with TH=NT (512 MB read once,
//   don't wash 192 MB L2), converted to bf16 on the fly; out stores TH=NT.
// ---------------------------------------------------------------------------

typedef __attribute__((ext_vector_type(16))) __bf16 v16bf;
typedef __attribute__((ext_vector_type(8)))  float  v8f;
typedef __attribute__((ext_vector_type(4)))  float  v4f;
typedef __attribute__((ext_vector_type(2)))  float  v2f;

#define T_STEPS 32
#define NNODES  2048
#define FEAT    64   // FI == FO == k

union Frag16 { v16bf v; unsigned int u[8]; };

__device__ __forceinline__ unsigned int pack_bf16x2(float a, float b) {
  unsigned int ua = __float_as_uint(a);
  unsigned int ub = __float_as_uint(b);
  ua = (ua + 0x7FFFu + ((ua >> 16) & 1u)) >> 16;   // RNE to bf16
  ub = (ub + 0x7FFFu + ((ub >> 16) & 1u)) >> 16;
  return (ua & 0xFFFFu) | (ub << 16);
}

__device__ __forceinline__ float sigmoidf_(float x) {
  return 1.0f / (1.0f + expf(-x));
}

#if defined(__has_builtin)
#if __has_builtin(__builtin_amdgcn_wmma_f32_16x16x4_f32)
#define HAS_WMMA_F32 1
#endif
#endif

#ifdef HAS_WMMA_F32
// One 16x16 output tile of (A1@B1 + A2@B2), all fp32 64x64 row-major operands.
// f32 WMMA layouts: A[m][k]: lane=m+16*(k>>1), vgpr=k&1.
//                   B[k][n]: lane=n+16*(k>>1), vgpr=k&1.
//                   C/D:     vgpr v -> M=v+8*(lane>>4), N=lane&15.
__device__ __forceinline__ v8f mm2_tile_f32(const float* A1, const float* B1,
                                            const float* A2, const float* B2,
                                            int ti, int tj, int lane) {
  v8f acc = {0.f, 0.f, 0.f, 0.f, 0.f, 0.f, 0.f, 0.f};
  const int m  = ti * 16 + (lane & 15);
  const int n  = tj * 16 + (lane & 15);
  const int ko = 2 * (lane >> 4);
#pragma unroll
  for (int k0 = 0; k0 < FEAT; k0 += 4) {
    v2f a, b;
    a.x = A1[m * FEAT + k0 + ko];
    a.y = A1[m * FEAT + k0 + ko + 1];
    b.x = B1[(k0 + ko) * FEAT + n];
    b.y = B1[(k0 + ko + 1) * FEAT + n];
    acc = __builtin_amdgcn_wmma_f32_16x16x4_f32(false, a, false, b,
                                                (short)0, acc, false, false);
  }
#pragma unroll
  for (int k0 = 0; k0 < FEAT; k0 += 4) {
    v2f a, b;
    a.x = A2[m * FEAT + k0 + ko];
    a.y = A2[m * FEAT + k0 + ko + 1];
    b.x = B2[(k0 + ko) * FEAT + n];
    b.y = B2[(k0 + ko + 1) * FEAT + n];
    acc = __builtin_amdgcn_wmma_f32_16x16x4_f32(false, a, false, b,
                                                (short)0, acc, false, false);
  }
  return acc;
}
#endif

// ---------------------------------------------------------------------------
// Phase 1: one block (1024 thr = 32 waves) per timestep; minimizes the length
// of the barrier-separated latency chain (bitonic substages do 1 compare-
// exchange per thread).
// ---------------------------------------------------------------------------
__global__ __launch_bounds__(1024) void egcn_topk_gru_kernel(
    const float* __restrict__ X,      // [N, FI]  nodes for step t
    const float* __restrict__ maskr,  // [N]
    const float* __restrict__ scorer, // [FI]
    const float* __restrict__ Wu, const float* __restrict__ Uu, const float* __restrict__ bu,
    const float* __restrict__ Wr, const float* __restrict__ Ur, const float* __restrict__ br,
    const float* __restrict__ Wh, const float* __restrict__ Uh, const float* __restrict__ bh,
    const float* __restrict__ Qprev,  // [64,64]
    float* __restrict__ Qnext)        // [64,64]
{
  __shared__ float sm0[4096];   // scores[2048] + idx[2048] during sort; later r / r*Q
  __shared__ float s_z[4096];   // z = (X[idx] * tanh(vals)).T  [FI][FO]
  __shared__ float s_u[4096];   // update gate
  __shared__ float s_tanh[FEAT];
  __shared__ int   s_sel[FEAT];

  const int tid = threadIdx.x;
  float* s_score = sm0;
  int*   s_idx   = ((int*)sm0) + 2048;

  // ||scorer|| (redundant per thread, 64 elems, L2-hot)
  float nrm2 = 0.f;
  for (int k = 0; k < FEAT; ++k) { float v = scorer[k]; nrm2 += v * v; }
  const float rnorm = 1.0f / sqrtf(nrm2);

  for (int n = tid; n < NNODES; n += 1024) {
    const float* xr = X + (size_t)n * FEAT;
    float d = 0.f;
    for (int k = 0; k < FEAT; ++k) d += xr[k] * scorer[k];
    s_score[n] = d * rnorm + maskr[n];
    s_idx[n]   = n;
  }

  // Bitonic sort, descending by score, ascending index on ties (top_k order)
  for (int k = 2; k <= NNODES; k <<= 1) {
    for (int j = k >> 1; j > 0; j >>= 1) {
      __syncthreads();
      for (int e = tid; e < NNODES; e += 1024) {
        int p = e ^ j;
        if (p > e) {
          float se = s_score[e], sp = s_score[p];
          int   ie = s_idx[e],   ip = s_idx[p];
          bool eFirst  = (se > sp) || (se == sp && ie < ip);
          bool descBlk = ((e & k) == 0);
          if (descBlk != eFirst) {
            s_score[e] = sp; s_score[p] = se;
            s_idx[e]   = ip; s_idx[p]   = ie;
          }
        }
      }
    }
  }
  __syncthreads();
  if (tid < FEAT) { s_sel[tid] = s_idx[tid]; s_tanh[tid] = tanhf(s_score[tid]); }
  __syncthreads();

  // z[f][j] = X[sel[j]][f] * tanh(val[j])
  for (int e = tid; e < FEAT * FEAT; e += 1024) {
    int f = e >> 6, j = e & 63;
    s_z[e] = X[(size_t)s_sel[j] * FEAT + f] * s_tanh[j];
  }
  __syncthreads();

  const int lane = tid & 31;
  const int wave = tid >> 5;

#ifdef HAS_WMMA_F32
  // u = sigmoid(Wu@z + Uu@Q + bu) ; r = sigmoid(Wr@z + Ur@Q + br)
  // 16 tiles of 16x16; waves 0..15 take one tile each (wave-uniform branch,
  // EXEC all-ones inside participating waves as WMMA requires).
  if (wave < 16) {
    const int ti = wave >> 2, tj = wave & 3;
    v8f au = mm2_tile_f32(Wu, s_z, Uu, Qprev, ti, tj, lane);
    v8f ar = mm2_tile_f32(Wr, s_z, Ur, Qprev, ti, tj, lane);
#pragma unroll
    for (int v = 0; v < 8; ++v) {
      int m = ti * 16 + v + ((lane >> 4) << 3);
      int n = tj * 16 + (lane & 15);
      s_u[m * FEAT + n] = sigmoidf_(au[v] + bu[m * FEAT + n]);
      sm0[m * FEAT + n] = sigmoidf_(ar[v] + br[m * FEAT + n]);
    }
  }
  __syncthreads();
  for (int e = tid; e < FEAT * FEAT; e += 1024) sm0[e] *= Qprev[e];   // r * Q
  __syncthreads();
  // h = relu(Wh@z + Uh@(r*Q) + bh); Qn = (1-u)*Q + u*h
  if (wave < 16) {
    const int ti = wave >> 2, tj = wave & 3;
    v8f ah = mm2_tile_f32(Wh, s_z, Uh, sm0, ti, tj, lane);
#pragma unroll
    for (int v = 0; v < 8; ++v) {
      int m = ti * 16 + v + ((lane >> 4) << 3);
      int n = tj * 16 + (lane & 15);
      float hh = fmaxf(ah[v] + bh[m * FEAT + n], 0.f);
      float uu = s_u[m * FEAT + n];
      float qq = Qprev[m * FEAT + n];
      Qnext[m * FEAT + n] = (1.0f - uu) * qq + uu * hh;
    }
  }
#else
  // VALU fallback
  for (int e = tid; e < FEAT * FEAT; e += 1024) {
    int m = e >> 6, n = e & 63;
    float su = bu[e], sr = br[e];
    for (int k = 0; k < FEAT; ++k) {
      float zz = s_z[k * FEAT + n];
      float qq = Qprev[k * FEAT + n];
      su = fmaf(Wu[m * FEAT + k], zz, fmaf(Uu[m * FEAT + k], qq, su));
      sr = fmaf(Wr[m * FEAT + k], zz, fmaf(Ur[m * FEAT + k], qq, sr));
    }
    s_u[e] = sigmoidf_(su);
    sm0[e] = sigmoidf_(sr);
  }
  __syncthreads();
  for (int e = tid; e < FEAT * FEAT; e += 1024) sm0[e] *= Qprev[e];
  __syncthreads();
  for (int e = tid; e < FEAT * FEAT; e += 1024) {
    int m = e >> 6, n = e & 63;
    float sh = bh[e];
    for (int k = 0; k < FEAT; ++k)
      sh = fmaf(Wh[m * FEAT + k], s_z[k * FEAT + n],
                fmaf(Uh[m * FEAT + k], sm0[k * FEAT + n], sh));
    float hh = fmaxf(sh, 0.f);
    float uu = s_u[e];
    Qnext[e] = (1.0f - uu) * Qprev[e] + uu * hh;
  }
#endif
}

// ---------------------------------------------------------------------------
// Phase 2: Y[t] = Nodes[t] @ Qn[t], output packed bf16 (32 dwords per row).
// ---------------------------------------------------------------------------
__global__ __launch_bounds__(256) void egcn_y_kernel(
    const float* __restrict__ Nodes,   // [T,N,FI]
    const float* __restrict__ Qn_all,  // [T,64,64]
    unsigned int* __restrict__ Y)      // [T,N,32] (bf16 pairs)
{
  __shared__ float sQ[FEAT * FEAT];
  const int t  = blockIdx.y;
  const int rb = blockIdx.x;           // 32 row-blocks of 64
  const float* Q = Qn_all + (size_t)t * FEAT * FEAT;
  for (int e = threadIdx.x; e < FEAT * FEAT; e += 256) sQ[e] = Q[e];
  __syncthreads();

  const int row = rb * 64 + (threadIdx.x >> 2);
  const int c0  = (threadIdx.x & 3) * 16;
  const float* x = Nodes + ((size_t)t * NNODES + row) * FEAT;
  float s[16];
#pragma unroll
  for (int c = 0; c < 16; ++c) s[c] = 0.f;
  for (int k = 0; k < FEAT; ++k) {
    float xv = x[k];
    const float* q = &sQ[k * FEAT + c0];
#pragma unroll
    for (int c = 0; c < 16; ++c) s[c] = fmaf(xv, q[c], s[c]);
  }
  unsigned int* yo = Y + ((size_t)t * NNODES + row) * 32 + (c0 >> 1);
#pragma unroll
  for (int c = 0; c < 8; ++c) yo[c] = pack_bf16x2(s[2 * c], s[2 * c + 1]);
}

// ---------------------------------------------------------------------------
// Phase 3: out[t] = relu(A[t] @ Y[t]).  HBM-bound stream of A (512 MB total),
// bf16 WMMA compute. Block: 8 waves x (16 rows x 64 cols) = 128x64 output.
// A loads and Out stores are non-temporal (streamed once); Y stays cached.
// ---------------------------------------------------------------------------
__global__ __launch_bounds__(256) void egcn_spmm_kernel(
    const float* __restrict__ A,        // [T,N,N]
    const unsigned int* __restrict__ Y, // [T,N,32] bf16 pairs
    float* __restrict__ Out)            // [T,N,64]
{
  __shared__ unsigned int ldsA[128 * 18];  // 128 rows x 32 bf16 (16 dwords) + pad
  __shared__ unsigned int ldsY[32 * 33];   // 32 K-rows x 64 bf16 (32 dwords) + pad

  const int t    = blockIdx.y;
  const int row0 = blockIdx.x * 128;
  const int tid  = threadIdx.x;
  const int lane = tid & 31;
  const int wave = tid >> 5;

  const int lrow = tid >> 1;   // 0..127 : A-tile row this thread loads
  const int seg  = tid & 1;    // 0..1   : 16-float segment within the 32-K chunk

  const float* Ab = A + (size_t)t * NNODES * NNODES
                      + (size_t)(row0 + lrow) * NNODES + seg * 16;
  const unsigned int* Yb = Y + (size_t)t * NNODES * 32;

  v8f acc0 = {0.f,0.f,0.f,0.f,0.f,0.f,0.f,0.f};
  v8f acc1 = {0.f,0.f,0.f,0.f,0.f,0.f,0.f,0.f};
  v8f acc2 = {0.f,0.f,0.f,0.f,0.f,0.f,0.f,0.f};
  v8f acc3 = {0.f,0.f,0.f,0.f,0.f,0.f,0.f,0.f};

  v4f p0, p1, p2, p3;
  unsigned int yp0, yp1, yp2, yp3;
  {
    const v4f* ap = (const v4f*)Ab;
    p0 = __builtin_nontemporal_load(ap);
    p1 = __builtin_nontemporal_load(ap + 1);
    p2 = __builtin_nontemporal_load(ap + 2);
    p3 = __builtin_nontemporal_load(ap + 3);
    yp0 = Yb[tid]; yp1 = Yb[tid + 256]; yp2 = Yb[tid + 512]; yp3 = Yb[tid + 768];
  }

  for (int k0 = 0; k0 < NNODES; k0 += 32) {
    __syncthreads();
    // stage A chunk (fp32 -> bf16) and Y tile into LDS
    unsigned int* da = &ldsA[lrow * 18 + seg * 8];
    da[0] = pack_bf16x2(p0.x, p0.y); da[1] = pack_bf16x2(p0.z, p0.w);
    da[2] = pack_bf16x2(p1.x, p1.y); da[3] = pack_bf16x2(p1.z, p1.w);
    da[4] = pack_bf16x2(p2.x, p2.y); da[5] = pack_bf16x2(p2.z, p2.w);
    da[6] = pack_bf16x2(p3.x, p3.y); da[7] = pack_bf16x2(p3.z, p3.w);
    {
      int i0 = tid;       ldsY[(i0 >> 5) * 33 + (i0 & 31)] = yp0;
      i0 = tid + 256;     ldsY[(i0 >> 5) * 33 + (i0 & 31)] = yp1;
      i0 = tid + 512;     ldsY[(i0 >> 5) * 33 + (i0 & 31)] = yp2;
      i0 = tid + 768;     ldsY[(i0 >> 5) * 33 + (i0 & 31)] = yp3;
    }
    __syncthreads();

    // prefetch next chunk while WMMAs run
    if (k0 + 32 < NNODES) {
      const v4f* ap = (const v4f*)(Ab + k0 + 32);
      p0 = __builtin_nontemporal_load(ap);
      p1 = __builtin_nontemporal_load(ap + 1);
      p2 = __builtin_nontemporal_load(ap + 2);
      p3 = __builtin_nontemporal_load(ap + 3);
      const unsigned int* yn = Yb + (size_t)(k0 + 32) * 32;
      yp0 = yn[tid]; yp1 = yn[tid + 256]; yp2 = yn[tid + 512]; yp3 = yn[tid + 768];
    }

    // A fragment: M = lane&15, K(dword) = 8*(i>>2) + 4*(lane>>4) + (i&3)
    Frag16 a;
    {
      const int rl = wave * 16 + (lane & 15);
      const int h  = lane >> 4;
#pragma unroll
      for (int i = 0; i < 8; ++i) {
        int kd = ((i >> 2) << 3) + (h << 2) + (i & 3);
        a.u[i] = ldsA[rl * 18 + kd];
      }
    }
    // B fragments: N = lane&15; VGPR i = K pair 2*(8*(i>>2) + 4*(lane>>4) + (i&3)) +{0,1}
    Frag16 b0, b1, b2, b3;
    {
      const int hb = lane >> 4;
      const int sh = (lane & 1) << 4;
#pragma unroll
      for (int i = 0; i < 8; ++i) {
        int kk = 2 * (((i >> 2) << 3) + (hb << 2) + (i & 3));
        const unsigned int* r0 = &ldsY[kk * 33 + ((lane & 15) >> 1)];
        const unsigned int* r1 = r0 + 33;
        b0.u[i] = ((r0[0] >> sh) & 0xFFFFu)  | ((r1[0] >> sh) << 16);
        b1.u[i] = ((r0[8] >> sh) & 0xFFFFu)  | ((r1[8] >> sh) << 16);
        b2.u[i] = ((r0[16] >> sh) & 0xFFFFu) | ((r1[16] >> sh) << 16);
        b3.u[i] = ((r0[24] >> sh) & 0xFFFFu) | ((r1[24] >> sh) << 16);
      }
    }
    acc0 = __builtin_amdgcn_wmma_f32_16x16x32_bf16(false, a.v, false, b0.v,
                                                   (short)0, acc0, false, false);
    acc1 = __builtin_amdgcn_wmma_f32_16x16x32_bf16(false, a.v, false, b1.v,
                                                   (short)0, acc1, false, false);
    acc2 = __builtin_amdgcn_wmma_f32_16x16x32_bf16(false, a.v, false, b2.v,
                                                   (short)0, acc2, false, false);
    acc3 = __builtin_amdgcn_wmma_f32_16x16x32_bf16(false, a.v, false, b3.v,
                                                   (short)0, acc3, false, false);
  }

  // writeback with relu: D vgpr v -> M = v + 8*(lane>>4), N = lane&15
  {
    const int mbase = row0 + wave * 16 + ((lane >> 4) << 3);
    const int n0    = lane & 15;
#pragma unroll
    for (int v = 0; v < 8; ++v) {
      float* orow = Out + ((size_t)t * NNODES + mbase + v) * 64;
      __builtin_nontemporal_store(fmaxf(acc0[v], 0.f), &orow[n0]);
      __builtin_nontemporal_store(fmaxf(acc1[v], 0.f), &orow[16 + n0]);
      __builtin_nontemporal_store(fmaxf(acc2[v], 0.f), &orow[32 + n0]);
      __builtin_nontemporal_store(fmaxf(acc3[v], 0.f), &orow[48 + n0]);
    }
  }
}

// ---------------------------------------------------------------------------
extern "C" void kernel_launch(void* const* d_in, const int* in_sizes, int n_in,
                              void* d_out, int out_size, void* d_ws, size_t ws_size,
                              hipStream_t stream) {
  (void)in_sizes; (void)n_in; (void)out_size; (void)ws_size;

  const float* A      = (const float*)d_in[0];   // [T,N,N]
  const float* Nodes  = (const float*)d_in[1];   // [T,N,FI]
  const float* Mask   = (const float*)d_in[2];   // [T,N]
  const float* Q0     = (const float*)d_in[3];   // [FI,FO]
  const float* scorer = (const float*)d_in[4];   // [FI,1]
  const float* Wu = (const float*)d_in[5];
  const float* Uu = (const float*)d_in[6];
  const float* bu = (const float*)d_in[7];
  const float* Wr = (const float*)d_in[8];
  const float* Ur = (const float*)d_in[9];
  const float* br = (const float*)d_in[10];
  const float* Wh = (const float*)d_in[11];
  const float* Uh = (const float*)d_in[12];
  const float* bh = (const float*)d_in[13];

  // workspace layout
  float*        Qn_all = (float*)d_ws;                                   // T*64*64 f32
  unsigned int* Ybf    = (unsigned int*)((char*)d_ws +
                          (size_t)T_STEPS * FEAT * FEAT * sizeof(float)); // T*N*32 u32 (bf16x2)

  // Phase 1: sequential Q-chain
  for (int t = 0; t < T_STEPS; ++t) {
    const float* Qprev = (t == 0) ? Q0 : (Qn_all + (size_t)(t - 1) * FEAT * FEAT);
    egcn_topk_gru_kernel<<<1, 1024, 0, stream>>>(
        Nodes + (size_t)t * NNODES * FEAT,
        Mask + (size_t)t * NNODES,
        scorer, Wu, Uu, bu, Wr, Ur, br, Wh, Uh, bh,
        Qprev, Qn_all + (size_t)t * FEAT * FEAT);
  }

  // Phase 2: Y[t] = X_t @ Qn_t (bf16 output), all timesteps in parallel
  egcn_y_kernel<<<dim3(NNODES / 64, T_STEPS), 256, 0, stream>>>(Nodes, Qn_all, Ybf);

  // Phase 3: out[t] = relu(A_t @ Y_t), bf16 WMMA, memory-bound on A stream
  egcn_spmm_kernel<<<dim3(NNODES / 128, T_STEPS), 256, 0, stream>>>(A, Ybf, (float*)d_out);
}
